// HierarchicalTreeLSTMs_82274393522870
// MI455X (gfx1250) — matrix operand order repaired and checked by reading, not compiled, not run on hardware
//
#include <hip/hip_runtime.h>
#include <math.h>
#include <stdint.h>

// ---------------- problem constants ----------------
#define N_NODES 4096
#define KCH     8
#define DDIM    512
#define HDIM    512
#define G4      2048          // 4*H gate width
#define SEQ     9             // 1 + K chain length
#define NBLK    32            // persistent blocks (must be co-resident)
#define NTHR    512           // 16 wave32s

typedef _Float16 f16;
typedef f16   v16h __attribute__((ext_vector_type(16)));
typedef f16   v8h  __attribute__((ext_vector_type(8)));
typedef float v8f  __attribute__((ext_vector_type(8)));

// Explicit global-address-space types: loads through these lower to
// global_load_b128 (LOADcnt only) instead of flat_load (LOADcnt+DScnt),
// decoupling the weight stream from the LDS pipe.
typedef f16 __attribute__((address_space(1))) ga_f16;
typedef v8h __attribute__((address_space(1))) ga_v8h;

union AFrag { v16h v; v8h h[2]; };

// f16 workspace layout (element offsets)
#define HALVES_W    (G4 * DDIM)                 // 1,048,576 per LSTM matrix
#define HALVES_ENC  (DDIM * 2 * HDIM)           // 524,288
#define HALVES_TOT  (4 * HALVES_W + HALVES_ENC) // 4,718,592
#define FLAGS_BYTE_OFF ((size_t)HALVES_TOT * 2) // flags after f16 region

__device__ __forceinline__ float sigmoidf_(float x) {
    return 1.0f / (1.0f + __expf(-x));
}

__device__ __forceinline__ v8f wmma_f16(v16h a, v16h b, v8f c) {
    return __builtin_amdgcn_wmma_f32_16x16x32_f16(false, a, false, b,
                                                  (short)0, c, false, false);
}

// Defeat LICM: make the weight pointer appear redefined here so the
// A-fragment loads cannot be hoisted out of enclosing loops (which caused
// a full preload + scratch spill). "+s" keeps it in SGPRs.
#define LAUNDER_PTR(p) asm volatile("" : "+s"(p))

__device__ __forceinline__ const ga_f16* to_global(const f16* p) {
    return (const ga_f16*)(uintptr_t)p;
}

// NT output tiles per wave over KLEN reduction. W is an explicit
// addrspace(1) pointer (laundered by the caller) so A-fragment fetches are
// global_load_b128 and stay inside the enclosing loops.
template<int NT, int KLEN>
__device__ __forceinline__ void gemv_tiles(const ga_f16* W,
                                           const f16* Bsrc,  // LDS row base
                                           int wave, int nIdx, int nHalf,
                                           v8f (&acc)[NT]) {
    #pragma unroll
    for (int tt = 0; tt < NT; ++tt) acc[tt] = (v8f){};
    #pragma unroll 1
    for (int kb = 0; kb < KLEN; kb += 32) {
        AFrag bf;
        bf.h[0] = *(const v8h*)(Bsrc + kb + 8 * nHalf);
        bf.h[1] = *(const v8h*)(Bsrc + kb + 16 + 8 * nHalf);
        v16h b = bf.v;
        #pragma unroll
        for (int tt = 0; tt < NT; ++tt) {
            const ga_f16* Wrow = W
                + (size_t)((wave * NT + tt) * 16 + nIdx) * KLEN
                + kb + 8 * nHalf;
            AFrag a;
            a.h[0] = *(const ga_v8h*)(Wrow);
            a.h[1] = *(const ga_v8h*)(Wrow + 16);
            acc[tt] = wmma_f16(a.v, b, acc[tt]);
        }
    }
}

// ------------- prep: f32 -> f16 weight copies + zero flags -------------
__global__ void prep_kernel(const float* __restrict__ Wl_ih,
                            const float* __restrict__ Wl_hh,
                            const float* __restrict__ Wr_ih,
                            const float* __restrict__ Wr_hh,
                            const float* __restrict__ W_enc,
                            f16* __restrict__ wsH,
                            unsigned* __restrict__ flags) {
    size_t i = (size_t)blockIdx.x * blockDim.x + threadIdx.x;
    size_t stride = (size_t)gridDim.x * blockDim.x;
    const size_t T = (size_t)HALVES_TOT;
    for (size_t x = i; x < T + N_NODES; x += stride) {
        if (x < T) {
            float v;
            if      (x <     HALVES_W) v = Wl_ih[x];
            else if (x < 2UL*HALVES_W) v = Wl_hh[x -     HALVES_W];
            else if (x < 3UL*HALVES_W) v = Wr_ih[x - 2UL*HALVES_W];
            else if (x < 4UL*HALVES_W) v = Wr_hh[x - 3UL*HALVES_W];
            else                       v = W_enc[x - 4UL*HALVES_W];
            wsH[x] = (f16)v;
        } else {
            flags[x - T] = 0u;
        }
    }
}

// ------------- main persistent TreeLSTM kernel -------------
__global__ __launch_bounds__(NTHR) void tree_lstm_kernel(
    const float* __restrict__ x0,
    const float* __restrict__ bl_ih, const float* __restrict__ bl_hh,
    const float* __restrict__ br_ih, const float* __restrict__ br_hh,
    const float* __restrict__ b_enc,
    const int*   __restrict__ lch,   const int* __restrict__ rch,
    const f16*   __restrict__ WihL,  const f16* __restrict__ WhhL,
    const f16*   __restrict__ WihR,  const f16* __restrict__ WhhR,
    const f16*   __restrict__ WencP,
    unsigned*    __restrict__ flags,
    float*       __restrict__ out)
{
    // ---- LDS (~143 KB of the 320 KB WGP budget) ----
    __shared__ __align__(32) f16   Xs[2][16][DDIM];   // chain inputs, t-major (rows 9..15 zero)
    __shared__ __align__(32) f16   Gx[2][SEQ][G4];    // precomputed input-gate terms
    __shared__ __align__(16) float Rr[2][G4];         // recurrent matvec result per step
    __shared__ __align__(32) f16   Hh[2][HDIM];       // h state (f16, contiguous => [hl;hr])
    __shared__ __align__(16) float Bsum[2][G4];       // b_ih + b_hh fused
    __shared__ __align__(16) float EncBuf[DDIM];
    __shared__ int ChildIdx[2][KCH];

    const int tid   = threadIdx.x;
    const int lane  = tid & 31;
    const int wave  = tid >> 5;       // 0..15
    const int nHalf = lane >> 4;      // 0/1 lane half-group
    const int nIdx  = lane & 15;      // B/D column (n), also A row (m)

    // fused biases (node-invariant), zero X padding rows once
    for (int j = tid; j < G4; j += NTHR) {
        Bsum[0][j] = bl_ih[j] + bl_hh[j];
        Bsum[1][j] = br_ih[j] + br_hh[j];
    }
    for (int s = 0; s < 2; ++s)
        for (int t = SEQ; t < 16; ++t)
            for (int d = tid; d < DDIM; d += NTHR) Xs[s][t][d] = (f16)0.f;
    __syncthreads();

    #pragma unroll 1
    for (int node = blockIdx.x; node < N_NODES; node += gridDim.x) {
        // ---- children + dependency wait ----
        if (tid < 2 * KCH) {
            int s = tid >> 3, k = tid & 7;
            ChildIdx[s][k] = (s == 0 ? lch : rch)[node * KCH + k];
        }
        __syncthreads();
        if (tid == 0) {
            for (int s = 0; s < 2; ++s)
                for (int k = 0; k < KCH; ++k) {
                    int c = ChildIdx[s][k];
                    if (c >= 0) {
                        while (__hip_atomic_load(&flags[c], __ATOMIC_RELAXED,
                                                 __HIP_MEMORY_SCOPE_AGENT) == 0u) {
                            __builtin_amdgcn_s_sleep(2);
                        }
                    }
                }
            __builtin_amdgcn_fence(__ATOMIC_ACQUIRE, "agent");
        }
        __syncthreads();

        // ---- build X (f16) and zero h ----
        for (int s = 0; s < 2; ++s) {
            for (int d = tid; d < DDIM; d += NTHR)
                Xs[s][0][d] = (f16)x0[(size_t)node * DDIM + d];
            for (int k = 0; k < KCH; ++k) {
                int c = ChildIdx[s][k];
                for (int d = tid; d < DDIM; d += NTHR)
                    Xs[s][k + 1][d] = (c >= 0) ? (f16)out[(size_t)c * DDIM + d]
                                               : (f16)0.f;
            }
        }
        for (int j = tid; j < HDIM; j += NTHR) {
            Hh[0][j] = (f16)0.f; Hh[1][j] = (f16)0.f;
        }
        __syncthreads();

        // ---- input-gate GEMM: Gx[s] = X[s] @ W_ih^T (WMMA, N = t batch) ----
        #pragma unroll
        for (int s = 0; s < 2; ++s) {
            const ga_f16* W = to_global((s == 0) ? WihL : WihR);
            LAUNDER_PTR(W);   // keep loads inside the node loop
            v8f acc[8];
            gemv_tiles<8, DDIM>(W, &Xs[s][nIdx][0], wave, nIdx, nHalf, acc);
            if (nIdx < SEQ) {
                #pragma unroll
                for (int tt = 0; tt < 8; ++tt) {
                    const int jt = wave * 8 + tt;
                    #pragma unroll
                    for (int v = 0; v < 8; ++v)
                        Gx[s][nIdx][jt * 16 + v + 8 * nHalf] = (f16)acc[tt][v];
                }
            }
        }
        __syncthreads();

        // ---- 9 recurrent steps ----
        float cL = 0.f, cR = 0.f;
        #pragma unroll 1
        for (int t = 0; t < SEQ; ++t) {
            // r[s] = W_hh[s] @ h[s]  (replicated-B WMMA GEMV)
            #pragma unroll
            for (int s = 0; s < 2; ++s) {
                const ga_f16* W = to_global((s == 0) ? WhhL : WhhR);
                LAUNDER_PTR(W);   // re-stream weights from L2 each step
                v8f acc[8];
                gemv_tiles<8, HDIM>(W, &Hh[s][0], wave, nIdx, nHalf, acc);
                if (nIdx == 0) {
                    #pragma unroll
                    for (int tt = 0; tt < 8; ++tt) {
                        const int jt = wave * 8 + tt;
                        #pragma unroll
                        for (int v = 0; v < 8; ++v)
                            Rr[s][jt * 16 + v + 8 * nHalf] = acc[tt][v];
                    }
                }
            }
            __syncthreads();
            // gating: one gate column per thread
            {
                const int j = tid;
                const bool mL = (t == 0) || (ChildIdx[0][t - 1] >= 0);
                const bool mR = (t == 0) || (ChildIdx[1][t - 1] >= 0);
                if (mL) {
                    float gi = (float)Gx[0][t][j]        + Rr[0][j]        + Bsum[0][j];
                    float gf = (float)Gx[0][t][j + 512]  + Rr[0][j + 512]  + Bsum[0][j + 512];
                    float gg = (float)Gx[0][t][j + 1024] + Rr[0][j + 1024] + Bsum[0][j + 1024];
                    float go = (float)Gx[0][t][j + 1536] + Rr[0][j + 1536] + Bsum[0][j + 1536];
                    float cn = sigmoidf_(gf) * cL + sigmoidf_(gi) * tanhf(gg);
                    cL = cn;
                    Hh[0][j] = (f16)(sigmoidf_(go) * tanhf(cn));
                }
                if (mR) {
                    float gi = (float)Gx[1][t][j]        + Rr[1][j]        + Bsum[1][j];
                    float gf = (float)Gx[1][t][j + 512]  + Rr[1][j + 512]  + Bsum[1][j + 512];
                    float gg = (float)Gx[1][t][j + 1024] + Rr[1][j + 1024] + Bsum[1][j + 1024];
                    float go = (float)Gx[1][t][j + 1536] + Rr[1][j + 1536] + Bsum[1][j + 1536];
                    float cn = sigmoidf_(gf) * cR + sigmoidf_(gi) * tanhf(gg);
                    cR = cn;
                    Hh[1][j] = (f16)(sigmoidf_(go) * tanhf(cn));
                }
            }
            __syncthreads();
        }

        // ---- encoder: enc = tanh(W_enc @ [hl;hr] + b_enc) ----
        {
            const ga_f16* W = to_global(WencP);
            LAUNDER_PTR(W);   // keep loads inside the node loop
            v8f acc[2];
            gemv_tiles<2, 2 * HDIM>(W, &Hh[0][0], wave, nIdx, nHalf, acc);
            if (nIdx == 0) {
                #pragma unroll
                for (int tt = 0; tt < 2; ++tt) {
                    const int jt = wave * 2 + tt;
                    #pragma unroll
                    for (int v = 0; v < 8; ++v)
                        EncBuf[jt * 16 + v + 8 * nHalf] = acc[tt][v];
                }
            }
        }
        __syncthreads();
        out[(size_t)node * DDIM + tid] = tanhf(EncBuf[tid] + b_enc[tid]);
        __syncthreads();
        if (tid == 0) {
            __builtin_amdgcn_fence(__ATOMIC_RELEASE, "agent");
            __hip_atomic_store(&flags[node], 1u, __ATOMIC_RELAXED,
                               __HIP_MEMORY_SCOPE_AGENT);
        }
        __syncthreads();
    }
}

// ------------- launch -------------
extern "C" void kernel_launch(void* const* d_in, const int* in_sizes, int n_in,
                              void* d_out, int out_size, void* d_ws, size_t ws_size,
                              hipStream_t stream) {
    const float* x0    = (const float*)d_in[0];
    const float* Wl_ih = (const float*)d_in[1];
    const float* Wl_hh = (const float*)d_in[2];
    const float* bl_ih = (const float*)d_in[3];
    const float* bl_hh = (const float*)d_in[4];
    const float* Wr_ih = (const float*)d_in[5];
    const float* Wr_hh = (const float*)d_in[6];
    const float* br_ih = (const float*)d_in[7];
    const float* br_hh = (const float*)d_in[8];
    const float* W_enc = (const float*)d_in[9];
    const float* b_enc = (const float*)d_in[10];
    const int*   lch   = (const int*)d_in[11];
    const int*   rch   = (const int*)d_in[12];

    f16*      wsH   = (f16*)d_ws;
    unsigned* flags = (unsigned*)((char*)d_ws + FLAGS_BYTE_OFF);
    float*    out   = (float*)d_out;

    const f16* WihL = wsH;
    const f16* WhhL = wsH + 1UL * HALVES_W;
    const f16* WihR = wsH + 2UL * HALVES_W;
    const f16* WhhR = wsH + 3UL * HALVES_W;
    const f16* Wenc = wsH + 4UL * HALVES_W;

    prep_kernel<<<2048, 256, 0, stream>>>(Wl_ih, Wl_hh, Wr_ih, Wr_hh, W_enc,
                                          wsH, flags);
    tree_lstm_kernel<<<NBLK, NTHR, 0, stream>>>(
        x0, bl_ih, bl_hh, br_ih, br_hh, b_enc, lch, rch,
        WihL, WhhL, WihR, WhhR, Wenc, flags, out);
}
